// Basic_GCN_layer_83373905150239
// MI455X (gfx1250) — matrix operand order repaired and checked by reading, not compiled, not run on hardware
//
#include <hip/hip_runtime.h>

// ---------------- problem constants ----------------
#define N_    32
#define CIN   64
#define COUT_ 64
#define T_    1024
#define V_    25
#define K_    3
#define OCH   (K_ * COUT_)     // 192
#define TB    8                // timesteps per block tile
#define MB    (TB * V_)        // 200 spatial positions per tile
#define MT    13               // ceil(MB/16) m-tiles (last ragged)
#define OT    (OCH / 16)       // 12 o-tiles
#define KP    96               // padded K for gemm2 (75 -> 96)
#define XS    216              // x tile LDS stride (f16 elems, mult of 8)
#define YS    104              // y2 LDS stride
#define AS    40               // Aflat LDS stride
#define CNT_F 819200.0f        // N*T*V
#define EPS_  1e-5f

typedef __attribute__((ext_vector_type(16))) _Float16 v16h;
typedef __attribute__((ext_vector_type(8)))  float    v8f;
typedef __attribute__((ext_vector_type(4)))  float    f4;

union FragH { v16h v; f4 q[2]; };

// ---------------- kernel 0: prep (f32 -> f16 weights, fused A+PA, zero accum) ----
__global__ __launch_bounds__(256) void gcn_prep(
    const float* __restrict__ W, const float* __restrict__ A,
    const float* __restrict__ PA, float* __restrict__ sums,
    _Float16* __restrict__ wh, _Float16* __restrict__ ah) {
  int tid = threadIdx.x;
  for (int i = tid; i < OCH * CIN; i += 256) wh[i] = (_Float16)W[i];
  for (int i = tid; i < KP * AS; i += 256) {
    int kk = i / AS, w = i - kk * AS;
    float val = 0.0f;
    if (kk < K_ * V_ && w < V_) val = A[kk * V_ + w] + PA[kk * V_ + w];
    ah[i] = (_Float16)val;   // stored [kk][w], w-contiguous
  }
  for (int i = tid; i < 128; i += 256) sums[i] = 0.0f;
}

// ---------------- kernel 1: fused conv-GEMM + graph-GEMM + BN partial sums ------
__global__ __launch_bounds__(256) void gcn_main(
    const float* __restrict__ x, const float* __restrict__ b,
    const _Float16* __restrict__ wh, const _Float16* __restrict__ ah,
    float* __restrict__ agg, float* __restrict__ sums) {
  extern __shared__ char smem[];
  _Float16* y2 = (_Float16*)smem;                    // [64*TB=512][YS]
  _Float16* xt = y2 + 512 * YS;                      // [CIN][XS]
  _Float16* wt = xt + CIN * XS;                      // [OCH][64]
  _Float16* at = wt + OCH * 64;                      // [KP][AS]
  float*    bs  = (float*)(at + KP * AS);            // [192] bias
  float*    acc = bs + OCH;                          // [128] BN partials
  float*    stage = (float*)smem;                    // f32 x staging (aliases y2)

  const int tid   = threadIdx.x;
  const int lane  = tid & 31;
  const int wave  = tid >> 5;
  const int laneN = lane & 15;       // column / row-within-16
  const int laneH = lane >> 4;       // which K-chunk half
  const int n  = blockIdx.x >> 7;    // batch
  const int t0 = (blockIdx.x & 127) * TB;

  // ---- phase 0: async DMA the raw f32 x tile into LDS staging (ASYNCcnt path),
  //      overlapped with regular staging of weights / adjacency / bias ----
  {
    // x tile rows: 64 channels x 200 floats = 50 16-byte chunks per row, all
    // 16B-aligned in both global and LDS.  GVS mode: mem = saddr + vaddr(i32).
    const unsigned gbase = (unsigned)(((n * CIN) * (T_ * V_) + t0 * V_) * 4);
    for (int i = tid; i < CIN * (MB / 4); i += 256) {   // 3200 chunks
      int c = i / (MB / 4), q = i - c * (MB / 4);
      unsigned lds_addr = (unsigned)(c * (MB * 4) + q * 16);         // into stage
      unsigned voff     = gbase + (unsigned)(c * (T_ * V_ * 4) + q * 16);
      asm volatile("global_load_async_to_lds_b128 %0, %1, %2"
                   :: "v"(lds_addr), "v"(voff), "s"(x)
                   : "memory");
    }
    const unsigned* ws32 = (const unsigned*)wh;
    unsigned* wd = (unsigned*)wt;
    for (int i = tid; i < OCH * CIN / 2; i += 256) wd[i] = ws32[i];
    const unsigned* as32 = (const unsigned*)ah;
    unsigned* ad = (unsigned*)at;
    for (int i = tid; i < KP * AS / 2; i += 256) ad[i] = as32[i];
    for (int i = tid; i < OCH; i += 256) bs[i] = b[i];
    for (int i = tid; i < 128; i += 256) acc[i] = 0.0f;
    asm volatile("s_wait_asynccnt 0x0" ::: "memory");
  }
  __syncthreads();

  // ---- phase 1: convert staged f32 tile -> f16 x-tile (staging dies here) ----
  for (int i = tid; i < CIN * MB; i += 256) {
    int c = i / MB, m = i - c * MB;
    xt[c * XS + m] = (_Float16)stage[c * MB + m];
  }
  __syncthreads();

  // ---- phase 2: GEMM1 y(192x200) = W(192x64) * xtile(64x200) + bias -> y2,
  //      plus zeroing of y2 pad columns kk in [75,96) (disjoint cells) ----
  for (int i = tid; i < 512 * (KP - K_ * V_); i += 256) {
    int r = i / 21, kk = (K_ * V_) + (i - r * 21);
    y2[r * YS + kk] = (_Float16)0.0f;
  }
  for (int tile = wave; tile < OT * MT; tile += 8) {
    int ot = tile / MT, mt = tile - ot * MT;
    v8f d = {};
#pragma unroll
    for (int kt = 0; kt < 2; ++kt) {
      FragH Af, Bf;
      const _Float16* ap = wt + (ot * 16 + laneN) * 64 + kt * 32 + laneH * 8;
      Af.q[0] = *(const f4*)(ap);
      Af.q[1] = *(const f4*)(ap + 16);
      const _Float16* bp = xt + (kt * 32 + laneH * 8) * XS + mt * 16 + laneN;
#pragma unroll
      for (int s = 0; s < 16; ++s)
        Bf.v[s] = bp[((s < 8) ? s : (s + 8)) * XS];
      d = __builtin_amdgcn_wmma_f32_16x16x32_f16(false, Af.v, false, Bf.v,
                                                 (short)0, d, false, false);
    }
    const int m = mt * 16 + laneN;
    if (m < MB) {
      const int tl = m / 25, vv = m - tl * 25;
#pragma unroll
      for (int r = 0; r < 8; ++r) {
        int o = ot * 16 + laneH * 8 + r;      // output channel 0..191
        int k = o >> 6, c = o & 63;
        y2[(c * TB + tl) * YS + k * V_ + vv] = (_Float16)(d[r] + bs[o]);
      }
    }
  }
  __syncthreads();

  // ---- phase 3: GEMM2 agg(512x25) = y2(512x96) * Aflat(96x32pad) + BN sums ----
  for (int tile = wave; tile < 64; tile += 8) {
    int rt = tile >> 1, ct = tile & 1;
    v8f d = {};
#pragma unroll
    for (int kt = 0; kt < 3; ++kt) {
      FragH Af, Bf;
      const _Float16* ap = y2 + (rt * 16 + laneN) * YS + kt * 32 + laneH * 8;
      Af.q[0] = *(const f4*)(ap);
      Af.q[1] = *(const f4*)(ap + 16);
      const _Float16* bp = at + (kt * 32 + laneH * 8) * AS + ct * 16 + laneN;
#pragma unroll
      for (int s = 0; s < 16; ++s)
        Bf.v[s] = bp[((s < 8) ? s : (s + 8)) * AS];
      d = __builtin_amdgcn_wmma_f32_16x16x32_f16(false, Af.v, false, Bf.v,
                                                 (short)0, d, false, false);
    }
    const int w = ct * 16 + laneN;
    const int c = rt * 2 + laneH;   // row R = rt*16 + laneH*8 + r ; c = R>>3 ; t_loc = r
    if (w < V_) {
      float s = 0.0f, s2 = 0.0f;
      const long obase = ((long)(n * 64 + c) * T_ + t0) * V_ + w;
#pragma unroll
      for (int r = 0; r < 8; ++r) {
        float val = d[r];
        agg[obase + (long)r * V_] = val;
        s += val;
        s2 += val * val;
      }
      atomicAdd(&acc[c], s);         // ds_add_f32
      atomicAdd(&acc[64 + c], s2);
    }
  }
  __syncthreads();
  for (int i = tid; i < 128; i += 256) atomicAdd(&sums[i], acc[i]);
}

// ---------------- kernel 2: finalize BN stats ----------------
__global__ void gcn_stats(const float* __restrict__ sums,
                          const float* __restrict__ gamma,
                          const float* __restrict__ beta,
                          float* __restrict__ stats) {
  int c = threadIdx.x;
  if (c < 64) {
    float mu  = sums[c] / CNT_F;
    float var = sums[64 + c] / CNT_F - mu * mu;
    float rs  = rsqrtf(var + EPS_);
    stats[c]       = mu;
    stats[64 + c]  = gamma[c] * rs;
    stats[128 + c] = beta[c];
  }
}

// ---------------- kernel 3: BN apply + residual + ReLU (vectorized) ----------------
__global__ __launch_bounds__(256) void gcn_apply(
    const f4* __restrict__ x4, f4* __restrict__ out4,
    const float* __restrict__ stats) {
  long i = (long)blockIdx.x * 256 + threadIdx.x;     // 13,107,200 exact
  int c = (int)((i / 6400) & 63);                    // 6400 float4 per (n,c) slab
  float mu = stats[c], gs = stats[64 + c], bt = stats[128 + c];
  f4 a = out4[i];
  f4 xv = x4[i];
  f4 r;
#pragma unroll
  for (int j = 0; j < 4; ++j) {
    float t = gs * (a[j] - mu) + bt + xv[j];
    r[j] = t > 0.0f ? t : 0.0f;
  }
  out4[i] = r;
}

// ---------------- host entry ----------------
extern "C" void kernel_launch(void* const* d_in, const int* in_sizes, int n_in,
                              void* d_out, int out_size, void* d_ws, size_t ws_size,
                              hipStream_t stream) {
  const float* x     = (const float*)d_in[0];
  const float* W     = (const float*)d_in[1];
  const float* b     = (const float*)d_in[2];
  const float* A     = (const float*)d_in[3];
  const float* PA    = (const float*)d_in[4];
  const float* gamma = (const float*)d_in[5];
  const float* beta  = (const float*)d_in[6];
  float* out = (float*)d_out;

  float* wsf      = (float*)d_ws;
  float* sums     = wsf;            // [128] sum / sumsq
  float* stats    = wsf + 128;      // [192] mu / gamma*rsig / beta
  _Float16* wh    = (_Float16*)(wsf + 320);   // [192*64]
  _Float16* ah    = wh + OCH * CIN;           // [96*40]

  gcn_prep<<<1, 256, 0, stream>>>(W, A, PA, sums, wh, ah);

  const size_t smem =
      (size_t)(512 * YS + CIN * XS + OCH * 64 + KP * AS) * sizeof(_Float16) +
      (OCH + 128) * sizeof(float);            // 167,680 B (2 blocks / WGP)
  gcn_main<<<N_ * (T_ / TB), 256, smem, stream>>>(x, b, wh, ah, out, sums);

  gcn_stats<<<1, 64, 0, stream>>>(sums, gamma, beta, stats);

  gcn_apply<<<(N_ * COUT_ * T_ * V_ / 4) / 256, 256, 0, stream>>>(
      (const f4*)x, (f4*)out, stats);
}
